// OctreeTracing_4080218931966
// MI455X (gfx1250) — matrix operand order repaired and checked by reading, not compiled, not run on hardware
//
#include <hip/hip_runtime.h>
#include <hip/hip_bf16.h>
#include <math.h>

typedef _Float16 half_t;
typedef __attribute__((ext_vector_type(16))) _Float16 v16h;
typedef __attribute__((ext_vector_type(8)))  _Float16 v8h;
typedef __attribute__((ext_vector_type(8)))  float    v8f;

#define N_TRACE   10
#define N_SECANT  8
#define SDF_THR   5e-5f

// LDS layout (bytes):
//   [0,      65536)  W1s (16384 f16, WMMA-B swizzled) + W2s (16384 f16)
//   [65536,  69120)  W0s(384 f32) b0s(128) b1s(128) b2s(128) W3s(128)
//   [69120, 135168)  8 per-wave regions of 8256B: actA(2048 f16) actB(2048 f16) sbuf(16 f32)
#define SMEM_BYTES   135168
#define WAVE_REGION  8256
#define WAVE_BASE    69120

__device__ __forceinline__ v8f wmma16(v16h a, v16h b, v8f c) {
  // v_wmma_f32_16x16x32_f16: D = A(16x32 f16) * B(32x16 f16) + C(16x16 f32)
  return __builtin_amdgcn_wmma_f32_16x16x32_f16(false, a, false, b, (short)0, c, false, false);
}

// A fragment for K-chunk kc from f16 row-major activations act[16][128].
// 16-bit A 16x32 layout: lanes 0-15 -> K {0..7, 16..23}, lanes 16-31 -> K {8..15, 24..31}.
__device__ __forceinline__ v16h load_a(const half_t* __restrict__ act, int m, int khalf, int kc) {
  const v8h lo = *(const v8h*)(act + m * 128 + kc * 32 + khalf);
  const v8h hi = *(const v8h*)(act + m * 128 + kc * 32 + 16 + khalf);
  return __builtin_shufflevector(lo, hi, 0,1,2,3,4,5,6,7,8,9,10,11,12,13,14,15);
}

// One neural-SDF evaluation for the wave's 16 rays; lane l evaluates ray m = l&15
// (lanes 16-31 mirror 0-15). Returns mlp(x) + b3 + |x| - 0.5 for the lane's ray.
__device__ __forceinline__ float sdf_eval(
    float x0, float x1, float x2, int lane,
    half_t* __restrict__ actA, half_t* __restrict__ actB, float* __restrict__ sbuf,
    const half_t* __restrict__ W1s, const half_t* __restrict__ W2s,
    const float* __restrict__ W0s, const float* __restrict__ b0s,
    const float* __restrict__ b1s, const float* __restrict__ b2s,
    const float* __restrict__ W3s, float b3v)
{
  const int m     = lane & 15;    // A-row / B-col index owned by this lane
  const int grp   = lane >> 4;    // 0: low half, 1: high half
  const int fbase = grp * 64;

  // ---- layer 0: x[3] @ W0[3,128] + b0, relu -> actA[16][128] f16 (VALU)
#pragma unroll
  for (int j0 = 0; j0 < 64; j0 += 8) {
    v8h pack;
#pragma unroll
    for (int e = 0; e < 8; ++e) {
      const int f = fbase + j0 + e;
      float h = fmaf(x0, W0s[f], fmaf(x1, W0s[128 + f], fmaf(x2, W0s[256 + f], b0s[f])));
      pack[e] = (half_t)fmaxf(h, 0.0f);
    }
    *(v8h*)(actA + m * 128 + fbase + j0) = pack;
  }

  const int khalf = grp * 8;
  const v16h* W1v = (const v16h*)W1s;
  const v16h* W2v = (const v16h*)W2s;

  // ---- layer 1: actA[16,128] @ W1[128,128] + b1, relu -> actB (WMMA)
  {
    v16h a0 = load_a(actA, m, khalf, 0);
    v16h a1 = load_a(actA, m, khalf, 1);
    v16h a2 = load_a(actA, m, khalf, 2);
    v16h a3 = load_a(actA, m, khalf, 3);
#pragma unroll
    for (int nt = 0; nt < 8; ++nt) {
      v8f acc = {0.f,0.f,0.f,0.f,0.f,0.f,0.f,0.f};
      acc = wmma16(a0, W1v[(nt * 4 + 0) * 32 + lane], acc);
      acc = wmma16(a1, W1v[(nt * 4 + 1) * 32 + lane], acc);
      acc = wmma16(a2, W1v[(nt * 4 + 2) * 32 + lane], acc);
      acc = wmma16(a3, W1v[(nt * 4 + 3) * 32 + lane], acc);
      const float bb = b1s[nt * 16 + m];
#pragma unroll
      for (int j = 0; j < 8; ++j) {
        const int row = grp ? (j + 8) : j;  // D layout: VGPR j -> row j / j+8
        actB[row * 128 + nt * 16 + m] = (half_t)fmaxf(acc[j] + bb, 0.0f);
      }
    }
  }

  // ---- layer 2 (WMMA) fused with final 128->1 dot against W3
  float p[8] = {0.f,0.f,0.f,0.f,0.f,0.f,0.f,0.f};
  {
    v16h a0 = load_a(actB, m, khalf, 0);
    v16h a1 = load_a(actB, m, khalf, 1);
    v16h a2 = load_a(actB, m, khalf, 2);
    v16h a3 = load_a(actB, m, khalf, 3);
#pragma unroll
    for (int nt = 0; nt < 8; ++nt) {
      v8f acc = {0.f,0.f,0.f,0.f,0.f,0.f,0.f,0.f};
      acc = wmma16(a0, W2v[(nt * 4 + 0) * 32 + lane], acc);
      acc = wmma16(a1, W2v[(nt * 4 + 1) * 32 + lane], acc);
      acc = wmma16(a2, W2v[(nt * 4 + 2) * 32 + lane], acc);
      acc = wmma16(a3, W2v[(nt * 4 + 3) * 32 + lane], acc);
      const float bb = b2s[nt * 16 + m];
      const float w3 = W3s[nt * 16 + m];
#pragma unroll
      for (int j = 0; j < 8; ++j)
        p[j] = fmaf(fmaxf(acc[j] + bb, 0.0f), w3, p[j]);
    }
  }

  // Reduce the dot over the 16 columns of each lane-half (wave32 butterflies).
#pragma unroll
  for (int j = 0; j < 8; ++j) {
    p[j] += __shfl_xor(p[j], 1, 32);
    p[j] += __shfl_xor(p[j], 2, 32);
    p[j] += __shfl_xor(p[j], 4, 32);
    p[j] += __shfl_xor(p[j], 8, 32);
  }
  if (lane == 0) {
#pragma unroll
    for (int j = 0; j < 8; ++j) sbuf[j] = p[j];       // rows 0..7
  }
  if (lane == 16) {
#pragma unroll
    for (int j = 0; j < 8; ++j) sbuf[8 + j] = p[j];   // rows 8..15
  }
  const float mlp = sbuf[m];  // same-wave LDS ordering guarantees visibility
  return mlp + b3v + sqrtf(fmaf(x0, x0, fmaf(x1, x1, x2 * x2))) - 0.5f;
}

__global__ __launch_bounds__(256) void octree_trace_kernel(
    const float* __restrict__ cam, const float* __restrict__ dirs,
    const float* __restrict__ W0g, const float* __restrict__ b0g,
    const float* __restrict__ W1g, const float* __restrict__ b1g,
    const float* __restrict__ W2g, const float* __restrict__ b2g,
    const float* __restrict__ W3g, const float* __restrict__ b3g,
    float* __restrict__ out_x, float* __restrict__ out_hit,
    float* __restrict__ out_t, int N)
{
  extern __shared__ char smem[];
  half_t* W1s = (half_t*)smem;
  half_t* W2s = W1s + 16384;
  float*  W0s = (float*)(smem + 65536);
  float*  b0s = W0s + 384;
  float*  b1s = b0s + 128;
  float*  b2s = b1s + 128;
  float*  W3s = b2s + 128;

  const int tid  = threadIdx.x;
  const int lane = tid & 31;
  const int wave = tid >> 5;
  char*   wb   = smem + WAVE_BASE + wave * WAVE_REGION;
  half_t* actA = (half_t*)wb;
  half_t* actB = actA + 2048;
  float*  sbuf = (float*)(wb + 8192);

  // Stage W1/W2 as f16 in WMMA B-fragment order:
  // frag(nt,kc) for lane l = W[k = 32*kc + 8*(l>>4) + (i&7) + 16*(i>>3)][n = 16*nt + (l&15)], i=0..15.
  for (int idx = tid; idx < 16384; idx += 256) {
    const int i  = idx & 15;
    const int ln = (idx >> 4) & 31;
    const int kc = (idx >> 9) & 3;
    const int nt = idx >> 11;
    const int k  = kc * 32 + (ln >> 4) * 8 + (i & 7) + (i >> 3) * 16;
    const int n  = nt * 16 + (ln & 15);
    W1s[idx] = (half_t)W1g[k * 128 + n];
    W2s[idx] = (half_t)W2g[k * 128 + n];
  }
  for (int i = tid; i < 384; i += 256) W0s[i] = W0g[i];
  for (int i = tid; i < 128; i += 256) {
    b0s[i] = b0g[i]; b1s[i] = b1g[i]; b2s[i] = b2g[i]; W3s[i] = W3g[i];
  }
  __syncthreads();

  const int m   = lane & 15;
  const int ray = blockIdx.x * 128 + wave * 16 + m;
  const int rr  = ray < N ? ray : N - 1;

  const float o0 = cam[0], o1 = cam[1], o2 = cam[2];
  const float d0 = dirs[rr * 3 + 0], d1 = dirs[rr * 3 + 1], d2 = dirs[rr * 3 + 2];
  const float b3v = b3g[0];

  // Ray / unit bounding-sphere intersection.
  const float bq   = o0 * d0 + o1 * d1 + o2 * d2;
  const float cq   = o0 * o0 + o1 * o1 + o2 * o2 - 1.0f;
  const float disc = bq * bq - cq;
  const bool  in_sphere = disc > 0.0f;
  const float sq    = sqrtf(fmaxf(disc, 0.0f));
  const float t_min = fmaxf(-bq - sq, 0.0f);
  const float t_max = fmaxf(-bq + sq, 0.0f);

  float t = t_min;
  float s = sdf_eval(o0 + t * d0, o1 + t * d1, o2 + t * d2, lane,
                     actA, actB, sbuf, W1s, W2s, W0s, b0s, b1s, b2s, W3s, b3v);

  float t_lo = 0.f, s_lo = 0.f, t_hi = 0.f, s_hi = 0.f;
  bool  br = false;

#pragma unroll 1
  for (int it = 0; it < N_TRACE; ++it) {
    const bool  go    = (!br) && (s > SDF_THR) && (t <= t_max) && in_sphere;
    const float t_new = go ? (t + s) : t;
    const float se    = sdf_eval(o0 + t_new * d0, o1 + t_new * d1, o2 + t_new * d2, lane,
                                 actA, actB, sbuf, W1s, W2s, W0s, b0s, b1s, b2s, W3s, b3v);
    const float s_new = go ? se : s;
    const bool  crossed = go && (s_new <= SDF_THR);
    t_lo = crossed ? t     : t_lo;  s_lo = crossed ? s     : s_lo;
    t_hi = crossed ? t_new : t_hi;  s_hi = crossed ? s_new : s_hi;
    br = br || crossed;
    t = t_new;  s = s_new;
  }

#pragma unroll 1
  for (int i = 0; i < N_SECANT; ++i) {
    const float denom = s_hi - s_lo;
    const float safe  = (fabsf(denom) > 1e-12f) ? denom : 1e-12f;
    float t_mid = t_lo - s_lo * (t_hi - t_lo) / safe;
    t_mid = fminf(fmaxf(t_mid, t_lo), t_hi);
    const float s_mid = sdf_eval(o0 + t_mid * d0, o1 + t_mid * d1, o2 + t_mid * d2, lane,
                                 actA, actB, sbuf, W1s, W2s, W0s, b0s, b1s, b2s, W3s, b3v);
    const bool pos = s_mid > 0.0f;
    t_lo = pos ? t_mid : t_lo;  s_lo = pos ? s_mid : s_lo;
    t_hi = pos ? t_hi : t_mid;  s_hi = pos ? s_hi : s_mid;
  }

  const float denom = s_hi - s_lo;
  const float safe  = (fabsf(denom) > 1e-12f) ? denom : 1e-12f;
  const float t_sec = fminf(fmaxf(t_lo - s_lo * (t_hi - t_lo) / safe, t_lo), t_hi);
  const float hit_t = br ? t_sec : t_max;

  if (lane < 16 && ray < N) {
    out_x[ray * 3 + 0] = o0 + hit_t * d0;
    out_x[ray * 3 + 1] = o1 + hit_t * d1;
    out_x[ray * 3 + 2] = o2 + hit_t * d2;
    out_hit[ray] = br ? 1.0f : 0.0f;
    out_t[ray]   = hit_t;
  }
}

extern "C" void kernel_launch(void* const* d_in, const int* in_sizes, int n_in,
                              void* d_out, int out_size, void* d_ws, size_t ws_size,
                              hipStream_t stream) {
  (void)n_in; (void)d_ws; (void)ws_size; (void)out_size;
  const float* cam  = (const float*)d_in[0];
  // d_in[1] = object_mask (unused by the reference computation)
  const float* dirs = (const float*)d_in[2];
  const float* W0   = (const float*)d_in[3];
  const float* b0   = (const float*)d_in[4];
  const float* W1   = (const float*)d_in[5];
  const float* b1   = (const float*)d_in[6];
  const float* W2   = (const float*)d_in[7];
  const float* b2   = (const float*)d_in[8];
  const float* W3   = (const float*)d_in[9];
  const float* b3   = (const float*)d_in[10];

  const int N = in_sizes[2] / 3;          // ray_directions is [1, N, 3]
  float* out   = (float*)d_out;
  float* out_x = out;                      // [N,3]
  float* out_h = out + (size_t)3 * N;      // [N] is_hit
  float* out_t = out + (size_t)4 * N;      // [N] hit_t

  const int grid = (N + 127) / 128;        // 128 rays per 256-thread block
  octree_trace_kernel<<<grid, 256, SMEM_BYTES, stream>>>(
      cam, dirs, W0, b0, W1, b1, W2, b2, W3, b3, out_x, out_h, out_t, N);
}
